// APPNP_28518582846060
// MI455X (gfx1250) — compile-verified, matrix operand order checked by reading, not run on hardware
//
#include <hip/hip_runtime.h>
#include <hip/hip_bf16.h>

// ---------------------------------------------------------------------------
// GAT (3-head) + APPNP(10) + fc head for MI455X (gfx1250, wave32, WMMA).
// Matrix stages: v_wmma_f32_16x16x32_bf16, fp32 accumulation, double-buffered
// LDS. A-operand staging: per-lane async global->LDS copies (ASYNCcnt) for the
// fp32-W GEMM, Tensor Data Mover descriptor loads (TENSORcnt) for bf16 GEMMs.
// ---------------------------------------------------------------------------

typedef __attribute__((ext_vector_type(16))) __bf16 v16bf;
typedef __attribute__((ext_vector_type(8)))  float  v8f;
typedef int vint4 __attribute__((vector_size(16)));        // builtin protos
typedef unsigned int vuint4 __attribute__((vector_size(16)));
typedef int vint8 __attribute__((vector_size(32)));

#define N_NODES 500
#define M_PAD   512
#define IN_FEAT 8192
#define OUT_FEAT 1024
#define N_HEADS 3
#define NHO     (N_HEADS * OUT_FEAT)   // 3072
#define K_STEPS 10
#define ALPHA   0.1f
#define NEG_SLOPE 0.2f
#define EPSV 1e-12f

static constexpr int KPAD = 48;  // LDS row stride (elements); 96B = 16B-aligned

// ---- CDNA5 async global->LDS copy (ASYNCcnt) ------------------------------
__device__ __forceinline__ void async_copy_b128(const void* g, void* l) {
#if __has_builtin(__builtin_amdgcn_global_load_async_to_lds_b128)
  __builtin_amdgcn_global_load_async_to_lds_b128(
      (__attribute__((address_space(1))) vint4*)g,
      (__attribute__((address_space(3))) vint4*)l, 0, 0);
#else
  asm volatile("global_load_async_to_lds_b128 %0, %1, off"
               :: "v"((unsigned)(unsigned long long)l), "v"(g)
               : "memory");
#endif
}

__device__ __forceinline__ void async_wait0() {
#if __has_builtin(__builtin_amdgcn_s_wait_asynccnt)
  __builtin_amdgcn_s_wait_asynccnt(0);
#else
  asm volatile("s_wait_asynccnt 0x0" ::: "memory");
#endif
}

__device__ __forceinline__ void tensor_wait0() {
#if __has_builtin(__builtin_amdgcn_s_wait_tensorcnt)
  __builtin_amdgcn_s_wait_tensorcnt(0);
#else
  asm volatile("s_wait_tensorcnt 0x0" ::: "memory");
#endif
}

// ---- CDNA5 Tensor Data Mover: load a 128x32 bf16 tile into padded LDS -----
// D# per CDNA5 ISA (2D, data_size=2B).  LDS padding:
// pad_interval code 3 (16 DWORDs = one 32-elem row), pad_amount code 7
// (8 DWORDs = 16 elems) -> LDS row stride 48 elements == KPAD.
__device__ __forceinline__ void tdm_load_tile_a(const __bf16* g, unsigned ldsoff,
                                                int lda) {
#if __has_builtin(__builtin_amdgcn_tensor_load_to_lds)
  const unsigned long long ga = (unsigned long long)g;
  const vuint4 g0 = {
      1u,                                         // count=1 (valid), user D#
      ldsoff,                                     // lds_addr (bytes)
      (unsigned)(ga & 0xFFFFFFFFu),               // global_addr[31:0]
      (unsigned)((ga >> 32) & 0x1FFFFFFu) | (2u << 30)  // addr[56:32] | type=2
  };
  const vint8 g1 = {
      (int)0x0ED10000u,        // dsize=2B<<16 | pad_en<<20 | intv=3<<22 | amt=7<<25
      (int)(32u << 16),        // [47:32] bar_addr=0, [63:48] tensor_dim0.lo = 32
      (int)(128u << 16),       // [79:64] tensor_dim0.hi=0, [95:80] tensor_dim1.lo=128
      (int)(32u << 16),        // [111:96] tensor_dim1.hi=0, [127:112] tile_dim0=32
      (int)128u,               // [143:128] tile_dim1=128, [159:144] tile_dim2=0
      lda,                     // [191:160] tensor_dim0_stride.lo (elements)
      0,                       // [207:192] stride0.hi, [223:208] stride1.lo=0
      0                        // stride1.hi
  };
  const vint4 z4 = {0, 0, 0, 0};
  const vint8 z8 = {0, 0, 0, 0, 0, 0, 0, 0};
  __builtin_amdgcn_tensor_load_to_lds(g0, g1, z4, z4, z8, 0);
#else
  (void)g; (void)ldsoff; (void)lda;
#endif
}

// ---------------------------------------------------------------------------
// Fragment load matching the CDNA5 16-bit A-matrix 16x32 lane layout:
//   lane<16 : row = row0+lane,     K = {0..7, 16..23}
//   lane>=16: row = row0+lane-16,  K = {8..15, 24..31}
// (B is staged into LDS K-transposed so the same loader produces B fragments.)
// ---------------------------------------------------------------------------
__device__ __forceinline__ v16bf load_frag(const __bf16* s, int row0, int lane) {
  const int r  = row0 + (lane & 15);
  const int kb = (lane >> 4) << 3;           // 0 or 8
  union { uint4 u[2]; v16bf v; } t;
  t.u[0] = *reinterpret_cast<const uint4*>(s + r * KPAD + kb);
  t.u[1] = *reinterpret_cast<const uint4*>(s + r * KPAD + kb + 16);
  return t.v;
}

// ---------------------------------------------------------------------------
// Generic bf16 WMMA GEMM:  C[Mpad x N] = s_acc * (A @ B) + s_add * addend
//   A: bf16, row-major (M x K), lda.  bf16-B build: staged by the TDM (one
//      tensor_load_to_lds per tile, wave 0).  fp32-B build: per-lane async
//      global->LDS b128 copies.
//   B: TB (float -> bf16 on the fly, or bf16), row-major (K x N), ldb
//   C: fp32, ldc; rows >= Mstore stored as 0 (keeps padding clean).
//   blockIdx.z selects a head via per-operand z-strides.
// Block tile 128x128, 8 waves (2x4), wave tile 64x32, BK=32, double-buffered.
// ---------------------------------------------------------------------------
template <typename TB>
__global__ __launch_bounds__(256)
void k_gemm(const __bf16* __restrict__ A, const TB* __restrict__ B,
            float* __restrict__ C, const float* __restrict__ addend,
            int K, int lda, int ldb, int ldc, int Mstore,
            float s_acc, float s_add,
            long long strideAz, long long strideBz, long long strideCz) {
  __shared__ __bf16 sA[2][128 * KPAD];
  __shared__ __bf16 sB[2][128 * KPAD];

  A += (size_t)blockIdx.z * strideAz;
  B += (size_t)blockIdx.z * strideBz;
  C += (size_t)blockIdx.z * strideCz;
  if (addend) addend += (size_t)blockIdx.z * strideCz;

  const int tid   = threadIdx.x;
  const int lane  = tid & 31;
  const int wid   = tid >> 5;
  const int wm    = wid >> 2;              // 0..1
  const int wn    = wid & 3;               // 0..3
  const int tileN = blockIdx.x * 128;
  const int tileM = blockIdx.y * 128;

  v8f acc[4][2] = {};

  const int arow = tid >> 1;               // 0..127
  const int akh  = (tid & 1) << 4;         // 0 or 16
  const int bkq  = (tid & 7) << 2;         // K sub-row group of 4: 0..28
  const int bcq  = (tid >> 3) << 2;        // col group of 4: 0..124

  // ---- A tile: 128 x 32 bf16 row-major into padded LDS --------------------
  auto stageA = [&](int buf, int kk) {
    if constexpr (sizeof(TB) == 2) {       // TDM descriptor load, wave 0 only
      if (wid == 0) {
        const __bf16* g = A + (size_t)tileM * lda + kk;
        const unsigned ldsoff =
            (unsigned)(unsigned long long)(__attribute__((address_space(3))) __bf16*)&sA[buf][0];
        tdm_load_tile_a(g, ldsoff, lda);
      }
    } else {                               // per-lane async b128 copies
      const __bf16* g = A + (size_t)(tileM + arow) * lda + kk + akh;
      __bf16* l = &sA[buf][arow * KPAD + akh];
      async_copy_b128(g, l);
      async_copy_b128(g + 8, l + 8);
    }
  };

  // ---- B tile: 32 x 128, K-transposed into sB[n][k]; 4x4 micro-tile/thread,
  //      packed ds_store_b64 per column (4 bf16 along K). -------------------
  auto stageB = [&](int buf, int kk) {
    __bf16 v4[4][4];                       // [k][n]
    if constexpr (sizeof(TB) == 4) {       // fp32 W: convert on the fly
#pragma unroll
      for (int r = 0; r < 4; ++r) {
        const float4 f = *reinterpret_cast<const float4*>(
            B + (size_t)(kk + bkq + r) * ldb + tileN + bcq);
        v4[r][0] = (__bf16)f.x; v4[r][1] = (__bf16)f.y;
        v4[r][2] = (__bf16)f.z; v4[r][3] = (__bf16)f.w;
      }
    } else {                               // already bf16
#pragma unroll
      for (int r = 0; r < 4; ++r) {
        union { uint2 u; __bf16 b[4]; } cv;
        cv.u = *reinterpret_cast<const uint2*>(
            B + (size_t)(kk + bkq + r) * ldb + tileN + bcq);
#pragma unroll
        for (int j = 0; j < 4; ++j) v4[r][j] = cv.b[j];
      }
    }
#pragma unroll
    for (int j = 0; j < 4; ++j) {
      union { uint2 u; __bf16 b[4]; } pk;
      pk.b[0] = v4[0][j]; pk.b[1] = v4[1][j];
      pk.b[2] = v4[2][j]; pk.b[3] = v4[3][j];
      *reinterpret_cast<uint2*>(&sB[buf][(bcq + j) * KPAD + bkq]) = pk.u;
    }
  };

  // Wait for this wave's outstanding staging traffic (ASYNCcnt or TENSORcnt).
  auto stage_wait = [&]() {
    if constexpr (sizeof(TB) == 2) {
      if (wid == 0) tensor_wait0();
    } else {
      async_wait0();
    }
  };

  // ---- software pipeline: prime buffer 0, then overlap copy/compute -------
  stageA(0, 0);
  stageB(0, 0);
  stage_wait();
  __syncthreads();

  int cur = 0;
  for (int kk = 0; kk < K; kk += 32) {
    const int nxt = cur ^ 1;
    if (kk + 32 < K) {                     // stage next slab while computing
      stageA(nxt, kk + 32);
      stageB(nxt, kk + 32);
      __builtin_prefetch(B + (size_t)(kk + 64 + bkq) * ldb + tileN + bcq, 0, 1);
    }

    v16bf af[4];
    v16bf bfrag[2];
#pragma unroll
    for (int fm = 0; fm < 4; ++fm)
      af[fm] = load_frag(sA[cur], wm * 64 + fm * 16, lane);
#pragma unroll
    for (int fn = 0; fn < 2; ++fn)
      bfrag[fn] = load_frag(sB[cur], wn * 32 + fn * 16, lane);

#pragma unroll
    for (int fm = 0; fm < 4; ++fm)
#pragma unroll
      for (int fn = 0; fn < 2; ++fn)
        acc[fm][fn] = __builtin_amdgcn_wmma_f32_16x16x32_bf16(
            false, af[fm], false, bfrag[fn], (short)0, acc[fm][fn], false, false);

    stage_wait();                          // next-buffer staging complete
    __syncthreads();                       // all waves done reading cur
    cur = nxt;
  }

  // ---- epilogue: C/D 16x16 fp32 layout (VGPR i -> M=i or i+8 by lane half)
#pragma unroll
  for (int fm = 0; fm < 4; ++fm) {
#pragma unroll
    for (int fn = 0; fn < 2; ++fn) {
      const int col = tileN + wn * 32 + fn * 16 + (lane & 15);
      const int rb  = tileM + wm * 64 + fm * 16 + ((lane >> 4) << 3);
#pragma unroll
      for (int i = 0; i < 8; ++i) {
        const int row    = rb + i;
        const size_t idx = (size_t)row * ldc + col;
        float v = s_acc * acc[fm][fn][i];
        if (addend) v += s_add * addend[idx];
        C[idx] = (row < Mstore) ? v : 0.0f;
      }
    }
  }
}

// ---------------------------------------------------------------------------
// Column L1 norms of x (sum over nodes, per feature), clamped at eps.
// ---------------------------------------------------------------------------
__global__ void k_colnorm(const float* __restrict__ x, float* __restrict__ cn) {
  const int f = blockIdx.x * 256 + threadIdx.x;
  float s = 0.0f;
  for (int i = 0; i < N_NODES; ++i) s += fabsf(x[(size_t)i * IN_FEAT + f]);
  cn[f] = fmaxf(s, EPSV);
}

// Normalize x and convert to bf16; pad rows [500,512) to zero.
__global__ void k_norm_x(const float* __restrict__ x, const float* __restrict__ cn,
                         __bf16* __restrict__ xb) {
  const int idx = blockIdx.x * 256 + threadIdx.x;   // M_PAD*IN_FEAT
  const int i = idx >> 13, f = idx & (IN_FEAT - 1);
  const float v = (i < N_NODES) ? x[(size_t)i * IN_FEAT + f] / cn[f] : 0.0f;
  xb[idx] = (__bf16)v;
}

// el[n,h] = h[n,h,:]·attn_l[h,:],  er likewise.
__global__ void k_el_er(const float* __restrict__ h, const float* __restrict__ al,
                        const float* __restrict__ ar, float* __restrict__ el,
                        float* __restrict__ er) {
  __shared__ float s1[128], s2[128];
  const int n = blockIdx.x, t = threadIdx.x;
  for (int hd = 0; hd < N_HEADS; ++hd) {
    float a = 0.0f, b = 0.0f;
    for (int o = t; o < OUT_FEAT; o += 128) {
      const float v = h[(size_t)n * NHO + hd * OUT_FEAT + o];
      a += v * al[hd * OUT_FEAT + o];
      b += v * ar[hd * OUT_FEAT + o];
    }
    s1[t] = a; s2[t] = b;
    __syncthreads();
    for (int st = 64; st > 0; st >>= 1) {
      if (t < st) { s1[t] += s1[t + st]; s2[t] += s2[t + st]; }
      __syncthreads();
    }
    if (t == 0) { el[n * N_HEADS + hd] = s1[0]; er[n * N_HEADS + hd] = s2[0]; }
    __syncthreads();
  }
}

// Masked leaky-ReLU softmax over sources; emits bf16 att[hd][i][j] (512x512 padded).
__global__ void k_attn(const int* __restrict__ A, const float* __restrict__ el,
                       const float* __restrict__ er, __bf16* __restrict__ att) {
  __shared__ float red[128];
  __shared__ float mglob, sglob;
  const int i = blockIdx.x, hd = blockIdx.y, t = threadIdx.x;
  __bf16* row = att + ((size_t)hd * M_PAD + i) * M_PAD;
  if (i >= N_NODES) {
    for (int j = t; j < M_PAD; j += 128) row[j] = (__bf16)0.0f;
    return;
  }
  const float eli = el[i * N_HEADS + hd];
  float m = -1e30f;
  for (int j = t; j < N_NODES; j += 128) {
    if (A[i * N_NODES + j] != 0) {
      float z = eli + er[j * N_HEADS + hd];
      z = (z > 0.0f) ? z : NEG_SLOPE * z;
      m = fmaxf(m, z);
    }
  }
  red[t] = m; __syncthreads();
  for (int st = 64; st > 0; st >>= 1) {
    if (t < st) red[t] = fmaxf(red[t], red[t + st]);
    __syncthreads();
  }
  if (t == 0) mglob = red[0];
  __syncthreads();
  const float mm = mglob;
  float s = 0.0f;
  for (int j = t; j < N_NODES; j += 128) {
    if (A[i * N_NODES + j] != 0) {
      float z = eli + er[j * N_HEADS + hd];
      z = (z > 0.0f) ? z : NEG_SLOPE * z;
      s += __expf(z - mm);
    }
  }
  red[t] = s; __syncthreads();
  for (int st = 64; st > 0; st >>= 1) {
    if (t < st) red[t] += red[t + st];
    __syncthreads();
  }
  if (t == 0) sglob = red[0];
  __syncthreads();
  const float inv = (sglob > 0.0f) ? 1.0f / sglob : 0.0f;
  for (int j = t; j < M_PAD; j += 128) {
    float v = 0.0f;
    if (j < N_NODES && A[i * N_NODES + j] != 0) {
      float z = eli + er[j * N_HEADS + hd];
      z = (z > 0.0f) ? z : NEG_SLOPE * z;
      v = __expf(z - mm) * inv;
    }
    row[j] = (__bf16)v;
  }
}

// dinv[i] = rsqrt(max(1 + rowsum(Af), eps))
__global__ void k_dinv(const int* __restrict__ A, float* __restrict__ dinv) {
  const int i = blockIdx.x * blockDim.x + threadIdx.x;
  if (i >= N_NODES) return;
  float s = 1.0f;
  for (int j = 0; j < N_NODES; ++j) s += (A[i * N_NODES + j] != 0) ? 1.0f : 0.0f;
  dinv[i] = rsqrtf(fmaxf(s, EPSV));
}

// Ahat (sym-normalized A + I) as bf16, zero-padded to 512x512.
__global__ void k_ahat(const int* __restrict__ A, const float* __restrict__ dinv,
                       __bf16* __restrict__ ah) {
  const int idx = blockIdx.x * 256 + threadIdx.x;   // M_PAD*M_PAD
  const int i = idx >> 9, j = idx & (M_PAD - 1);
  float v = 0.0f;
  if (i < N_NODES && j < N_NODES) {
    float a = (A[i * N_NODES + j] != 0) ? 1.0f : 0.0f;
    if (i == j) a += 1.0f;
    v = a * dinv[i] * dinv[j];
  }
  ah[idx] = (__bf16)v;
}

// fp32 -> bf16 element convert.
__global__ void k_cvt(const float* __restrict__ in, __bf16* __restrict__ outp, int n) {
  const int idx = blockIdx.x * 256 + threadIdx.x;
  if (idx < n) outp[idx] = (__bf16)in[idx];
}

// fc: deterministic two-stage reduction of fc_w @ flat(hk).
__global__ void k_fc_part(const float* __restrict__ hk, const float* __restrict__ fw,
                          float* __restrict__ part) {
  __shared__ float r0[256], r1[256];
  const int t = threadIdx.x;
  const size_t base = (size_t)blockIdx.x * 1024;
  float p0 = 0.0f, p1 = 0.0f;
  for (int q = t; q < 1024; q += 256) {
    const float v = hk[base + q];
    p0 += fw[base + q] * v;
    p1 += fw[(size_t)N_NODES * NHO + base + q] * v;
  }
  r0[t] = p0; r1[t] = p1; __syncthreads();
  for (int st = 128; st > 0; st >>= 1) {
    if (t < st) { r0[t] += r0[t + st]; r1[t] += r1[t + st]; }
    __syncthreads();
  }
  if (t == 0) { part[blockIdx.x * 2] = r0[0]; part[blockIdx.x * 2 + 1] = r1[0]; }
}

__global__ void k_fc_final(const float* __restrict__ part, const float* __restrict__ fb,
                           float* __restrict__ outp) {
  __shared__ float r0[256], r1[256];
  const int t = threadIdx.x;
  float p0 = 0.0f, p1 = 0.0f;
  for (int i = t; i < 1500; i += 256) { p0 += part[i * 2]; p1 += part[i * 2 + 1]; }
  r0[t] = p0; r1[t] = p1; __syncthreads();
  for (int st = 128; st > 0; st >>= 1) {
    if (t < st) { r0[t] += r0[t + st]; r1[t] += r1[t + st]; }
    __syncthreads();
  }
  if (t == 0) { outp[0] = r0[0] + fb[0]; outp[1] = r1[0] + fb[1]; }
}

// ---------------------------------------------------------------------------
extern "C" void kernel_launch(void* const* d_in, const int* in_sizes, int n_in,
                              void* d_out, int out_size, void* d_ws, size_t ws_size,
                              hipStream_t stream) {
  (void)in_sizes; (void)n_in; (void)out_size; (void)ws_size;
  const int*   A      = (const int*)  d_in[0];
  const float* x      = (const float*)d_in[1];
  const float* W      = (const float*)d_in[2];
  const float* attn_l = (const float*)d_in[3];
  const float* attn_r = (const float*)d_in[4];
  const float* fc_w   = (const float*)d_in[5];
  const float* fc_b   = (const float*)d_in[6];
  float* out = (float*)d_out;

  char* p = (char*)d_ws;
  size_t off = 0;
  auto take = [&](size_t bytes) -> void* {
    void* r = p + off;
    off += (bytes + 255) & ~(size_t)255;
    return r;
  };
  __bf16* xb     = (__bf16*)take((size_t)M_PAD * IN_FEAT * 2);  // normalized x, bf16
  float*  h      = (float*) take((size_t)M_PAD * NHO * 4);      // per-head projections
  __bf16* hbf    = (__bf16*)take((size_t)M_PAD * NHO * 2);      // bf16 scratch (h / hc)
  float*  out0   = (float*) take((size_t)M_PAD * NHO * 4);      // h0 for APPNP teleport
  float*  buf1   = (float*) take((size_t)M_PAD * NHO * 4);
  float*  buf2   = (float*) take((size_t)M_PAD * NHO * 4);
  __bf16* attbf  = (__bf16*)take((size_t)N_HEADS * M_PAD * M_PAD * 2);
  __bf16* ahatbf = (__bf16*)take((size_t)M_PAD * M_PAD * 2);
  float*  cn     = (float*) take(IN_FEAT * 4);
  float*  el     = (float*) take(1536 * 4);
  float*  er     = (float*) take(1536 * 4);
  float*  dinv   = (float*) take(M_PAD * 4);
  float*  part   = (float*) take(3000 * 4);

  // 1) column L1 normalize x -> bf16 (padded to 512 rows)
  k_colnorm<<<IN_FEAT / 256, 256, 0, stream>>>(x, cn);
  k_norm_x<<<(M_PAD * IN_FEAT) / 256, 256, 0, stream>>>(x, cn, xb);

  // 2) h[:, hd, :] = xb @ W[hd]  (all 3 heads in one launch via blockIdx.z)
  k_gemm<float><<<dim3(OUT_FEAT / 128, M_PAD / 128, N_HEADS), 256, 0, stream>>>(
      xb, W, h, nullptr, IN_FEAT, IN_FEAT, OUT_FEAT, NHO, N_NODES, 1.0f, 0.0f,
      0LL, (long long)IN_FEAT * OUT_FEAT, (long long)OUT_FEAT);

  // 3) attention scores and masked softmax
  k_el_er<<<N_NODES, 128, 0, stream>>>(h, attn_l, attn_r, el, er);
  k_attn<<<dim3(M_PAD, N_HEADS), 128, 0, stream>>>(A, el, er, attbf);

  // 4) out0[:, hd, :] = att[hd] @ h[:, hd, :]   (A tiles staged by the TDM)
  k_cvt<<<(M_PAD * NHO) / 256, 256, 0, stream>>>(h, hbf, M_PAD * NHO);
  k_gemm<__bf16><<<dim3(OUT_FEAT / 128, M_PAD / 128, N_HEADS), 256, 0, stream>>>(
      attbf, hbf, out0, nullptr, M_PAD, M_PAD, NHO, NHO, N_NODES, 1.0f, 0.0f,
      (long long)M_PAD * M_PAD, (long long)OUT_FEAT, (long long)OUT_FEAT);

  // 5) APPNP: hk = 0.9 * (Ahat @ hk) + 0.1 * h0, 10 steps (TDM-staged A)
  k_dinv<<<2, 256, 0, stream>>>(A, dinv);
  k_ahat<<<(M_PAD * M_PAD) / 256, 256, 0, stream>>>(A, dinv, ahatbf);
  float* cur = out0;
  float* nxt = buf1;
  for (int k = 0; k < K_STEPS; ++k) {
    k_cvt<<<(M_PAD * NHO) / 256, 256, 0, stream>>>(cur, hbf, M_PAD * NHO);
    k_gemm<__bf16><<<dim3(NHO / 128, M_PAD / 128, 1), 256, 0, stream>>>(
        ahatbf, hbf, nxt, out0, M_PAD, M_PAD, NHO, NHO, N_NODES,
        1.0f - ALPHA, ALPHA, 0LL, 0LL, 0LL);
    cur = nxt;
    nxt = (cur == buf1) ? buf2 : buf1;
  }

  // 6) fc head (deterministic two-stage reduction)
  k_fc_part<<<(N_NODES * NHO) / 1024, 256, 0, stream>>>(cur, fc_w, part);
  k_fc_final<<<1, 256, 0, stream>>>(part, fc_b, out);
}
